// IpuTFBertSelfAttention_68461778698695
// MI455X (gfx1250) — matrix operand order repaired and checked by reading, not compiled, not run on hardware
//
#include <hip/hip_runtime.h>
#include <hip/hip_bf16.h>

// ---------------------------------------------------------------------------
// BERT self-attention for gfx1250 (MI455X), wave32 + V_WMMA_F32_16X16X32_BF16
// + GLOBAL_LOAD_ASYNC_TO_LDS_B128 cooperative staging.
// B=8, S=1024, D=1024, H=16, DH=64.
// ---------------------------------------------------------------------------

#define Bb 8
#define Ss 1024
#define Dd 1024
#define Hh 16
#define DH 64
#define N3D 3072

typedef __attribute__((ext_vector_type(16))) __bf16 v16bf;
typedef __attribute__((ext_vector_type(8)))  float  v8f;
typedef int v4i_gcc __attribute__((vector_size(16)));

#define AS1 __attribute__((address_space(1)))
#define AS3 __attribute__((address_space(3)))

union V16BF {
    v16bf          v;
    uint4          q[2];
    unsigned short s[16];
};

static __device__ __forceinline__ unsigned short f32_to_bf16(float f) {
    unsigned int u = __float_as_uint(f);
    u += 0x7FFFu + ((u >> 16) & 1u);   // round to nearest even
    return (unsigned short)(u >> 16);
}

static __device__ __forceinline__ v8f vzero8() {
    v8f z;
#pragma unroll
    for (int i = 0; i < 8; ++i) z[i] = 0.0f;
    return z;
}

// 16-byte async copy global -> LDS (per-lane addresses), ASYNCcnt-tracked.
static __device__ __forceinline__ void async16(const void* g, void* l) {
    __builtin_amdgcn_global_load_async_to_lds_b128(
        (AS1 v4i_gcc*)(unsigned long long)g,
        (AS3 v4i_gcc*)(unsigned int)(unsigned long long)l, 0, 0);
}

// ---------------------------------------------------------------------------
// Kernel 1: hidden_states fp32 -> bf16, same [B*S, D] layout.
// ---------------------------------------------------------------------------
__global__ void k_cvt_hidden(const float* __restrict__ src,
                             unsigned short* __restrict__ dst) {
    int idx = blockIdx.x * 256 + threadIdx.x;   // 4 elements per thread
    const float4 f = ((const float4*)src)[idx];
    ushort4 o;
    o.x = f32_to_bf16(f.x);
    o.y = f32_to_bf16(f.y);
    o.z = f32_to_bf16(f.z);
    o.w = f32_to_bf16(f.w);
    ((ushort4*)dst)[idx] = o;
}

// ---------------------------------------------------------------------------
// Kernel 2: weight [K=1024, N=3072] fp32 -> Wt [N=3072, K=1024] bf16.
// LDS-tiled 32x32 transpose; grid = (K/32, N/32).
// ---------------------------------------------------------------------------
__global__ void k_cvt_weight(const float* __restrict__ w,
                             unsigned short* __restrict__ wt) {
    __shared__ float tile[32][33];
    const int tx = threadIdx.x & 31;
    const int ty = threadIdx.x >> 5;           // 0..7
    const int k0 = blockIdx.x * 32;
    const int n0 = blockIdx.y * 32;
#pragma unroll
    for (int i = 0; i < 4; ++i) {
        int k = k0 + ty + i * 8;
        tile[ty + i * 8][tx] = w[k * N3D + (n0 + tx)];
    }
    __syncthreads();
#pragma unroll
    for (int i = 0; i < 4; ++i) {
        int n = n0 + ty + i * 8;
        wt[n * Dd + (k0 + tx)] = f32_to_bf16(tile[tx][ty + i * 8]);
    }
}

// ---------------------------------------------------------------------------
// Kernel 3: QKV GEMM, bf16 in / fp32 acc / bf16 out. 32x64 tile per wave:
// 2 A-frags x 4 B-frags -> 8 WMMAs per k-step (3x fragment reuse).
// Each wave's 64-wide n-tile = exactly one (proj, head), dh 0..63.
//   Q,K out: [B,H,S,DH] bf16      V out (transposed): [B,H,DH,S] bf16
// ---------------------------------------------------------------------------
__global__ void k_qkv_gemm(const unsigned short* __restrict__ hid,
                           const unsigned short* __restrict__ wt,
                           unsigned short* __restrict__ qo,
                           unsigned short* __restrict__ ko,
                           unsigned short* __restrict__ vo) {
    const int gw   = blockIdx.x * 8 + (threadIdx.x >> 5);  // 0..12287
    const int lane = threadIdx.x & 31;
    const int lm   = lane & 15;
    const int hi   = lane >> 4;

    const int tm = gw / (N3D / 64);   // 0..255  (rows of 32)
    const int tn = gw % (N3D / 64);   // 0..47   (cols of 64)
    const int m0 = tm * 32;
    const int n0 = tn * 64;

    const unsigned short* arow0 = hid + (m0 + lm) * Dd;
    const unsigned short* arow1 = hid + (m0 + 16 + lm) * Dd;
    const unsigned short* brow0 = wt + (n0 + 0 * 16 + lm) * Dd;
    const unsigned short* brow1 = wt + (n0 + 1 * 16 + lm) * Dd;
    const unsigned short* brow2 = wt + (n0 + 2 * 16 + lm) * Dd;
    const unsigned short* brow3 = wt + (n0 + 3 * 16 + lm) * Dd;

    v8f acc[2][4];
#pragma unroll
    for (int jm = 0; jm < 2; ++jm)
#pragma unroll
        for (int jn = 0; jn < 4; ++jn) acc[jm][jn] = vzero8();

#pragma unroll 2
    for (int kk = 0; kk < Dd; kk += 32) {
        V16BF a[2], b[4];
        a[0].q[0] = *(const uint4*)(arow0 + kk + 16 * hi);
        a[0].q[1] = *(const uint4*)(arow0 + kk + 16 * hi + 8);
        a[1].q[0] = *(const uint4*)(arow1 + kk + 16 * hi);
        a[1].q[1] = *(const uint4*)(arow1 + kk + 16 * hi + 8);
        const uint4* bp;
        bp = (const uint4*)(brow0 + kk + 16 * hi); b[0].q[0] = bp[0]; b[0].q[1] = bp[1];
        bp = (const uint4*)(brow1 + kk + 16 * hi); b[1].q[0] = bp[0]; b[1].q[1] = bp[1];
        bp = (const uint4*)(brow2 + kk + 16 * hi); b[2].q[0] = bp[0]; b[2].q[1] = bp[1];
        bp = (const uint4*)(brow3 + kk + 16 * hi); b[3].q[0] = bp[0]; b[3].q[1] = bp[1];
#pragma unroll
        for (int jm = 0; jm < 2; ++jm)
#pragma unroll
            for (int jn = 0; jn < 4; ++jn)
                acc[jm][jn] = __builtin_amdgcn_wmma_f32_16x16x32_bf16(
                    false, a[jm].v, false, b[jn].v, (short)0, acc[jm][jn],
                    false, false);
    }

    // n-tile = one (proj, head); dh = jn*16 + lm
    const int proj = n0 >> 10;
    const int h    = (n0 & 1023) >> 6;
    const int bb   = m0 >> 10;
    const int s0   = m0 & 1023;
    const int bh   = bb * Hh + h;

    if (proj == 2) {                       // V transposed: [B,H,DH,S]
#pragma unroll
        for (int jn = 0; jn < 4; ++jn) {
            unsigned short* base = vo + (bh * DH + jn * 16 + lm) * Ss + s0;
#pragma unroll
            for (int jm = 0; jm < 2; ++jm)
#pragma unroll
                for (int r = 0; r < 8; ++r)
                    base[jm * 16 + r + 8 * hi] = f32_to_bf16(acc[jm][jn][r]);
        }
    } else {                               // Q or K: [B,H,S,DH]
        unsigned short* dst = (proj == 0) ? qo : ko;
#pragma unroll
        for (int jm = 0; jm < 2; ++jm)
#pragma unroll
            for (int r = 0; r < 8; ++r) {
                unsigned short* rowp =
                    dst + (bh * Ss + s0 + jm * 16 + r + 8 * hi) * DH + lm;
#pragma unroll
                for (int jn = 0; jn < 4; ++jn)
                    rowp[jn * 16] = f32_to_bf16(acc[jm][jn][r]);
            }
    }
}

// ---------------------------------------------------------------------------
// Kernel 4: fused flash attention. Block = 8 consecutive q-tiles of ONE head.
// Each 32-key chunk of K and V(T) is async-copied to LDS once per block and
// shared by all 8 waves (8x traffic reduction vs per-wave streaming).
// ---------------------------------------------------------------------------
#define WAVE_LDS 640   // P 16*33 + rmax16 + rsum16 + rscale16 + pmax32 + psum32

__global__ void k_attention(const unsigned short* __restrict__ qb,
                            const unsigned short* __restrict__ kb,
                            const unsigned short* __restrict__ vtb,
                            const float* __restrict__ mask,
                            float* __restrict__ out) {
    __shared__ __align__(16) unsigned short Ks[32 * DH];   // [key32][dh64]
    __shared__ __align__(16) unsigned short Vs[DH * 32];   // [dh64][key32]
    __shared__ float smem[8 * WAVE_LDS];

    const int t    = threadIdx.x;
    const int wave = t >> 5;
    const int lane = t & 31;
    const int lm   = lane & 15;
    const int hi   = lane >> 4;

    float* P      = smem + wave * WAVE_LDS;   // [16][33]
    float* rmax   = P + 528;
    float* rsum   = P + 544;
    float* rscale = P + 560;
    float* pmax   = P + 576;                  // [2][16]
    float* psum   = P + 608;                  // [2][16]

    const int bh = blockIdx.x >> 3;                    // head id 0..127
    const int qt = (blockIdx.x & 7) * 8 + wave;        // q-tile 0..63
    const int bb = bh >> 4;
    const int h  = bh & 15;
    const int q0 = qt * 16;

    const unsigned short* qhead = qb  + bh * Ss * DH;
    const unsigned short* khead = kb  + bh * Ss * DH;
    const unsigned short* vhead = vtb + bh * DH * Ss;
    const float*          mrow  = mask + bb * Ss;

    // Q A-fragments (dh 0..31 and 32..63), in registers for whole loop.
    V16BF qa0, qa1;
    {
        const unsigned short* qrow = qhead + (q0 + lm) * DH;
        qa0.q[0] = *(const uint4*)(qrow + 16 * hi);
        qa0.q[1] = *(const uint4*)(qrow + 16 * hi + 8);
        qa1.q[0] = *(const uint4*)(qrow + 32 + 16 * hi);
        qa1.q[1] = *(const uint4*)(qrow + 32 + 16 * hi + 8);
    }

    if (lane < 16) { rmax[lane] = -1e30f; rsum[lane] = 0.0f; }

    v8f o0 = vzero8(), o1 = vzero8(), o2 = vzero8(), o3 = vzero8();

    for (int kt = 0; kt < Ss / 32; ++kt) {
        const int key0 = kt * 32;

        // ---- cooperative async stage of K (4KB) and Vt (4KB) chunk ----
        {
            const int kr = t >> 3, kp = t & 7;         // 32 rows x 8 xb128
            async16(khead + (key0 + kr) * DH + kp * 8, &Ks[kr * DH + kp * 8]);
            const int vr = t >> 2, vp = t & 3;         // 64 rows x 4 xb128
            async16(vhead + vr * Ss + key0 + vp * 8, &Vs[vr * 32 + vp * 8]);
        }
        asm volatile("s_wait_asynccnt 0" ::: "memory");
        __syncthreads();

        // ---- scores: two 16-key tiles, K-dim = DH (2 wmma each) ----
        v8f sc[2];
#pragma unroll
        for (int j = 0; j < 2; ++j) {
            sc[j] = vzero8();
            const unsigned short* krow = &Ks[(j * 16 + lm) * DH];
            V16BF b0, b1;
            const uint4* p0 = (const uint4*)(krow + 16 * hi);
            b0.q[0] = p0[0]; b0.q[1] = p0[1];
            const uint4* p1 = (const uint4*)(krow + 32 + 16 * hi);
            b1.q[0] = p1[0]; b1.q[1] = p1[1];
            sc[j] = __builtin_amdgcn_wmma_f32_16x16x32_bf16(
                        false, qa0.v, false, b0.v, (short)0, sc[j], false, false);
            sc[j] = __builtin_amdgcn_wmma_f32_16x16x32_bf16(
                        false, qa1.v, false, b1.v, (short)0, sc[j], false, false);
        }

        // ---- scale + additive mask, spill 16x32 tile to LDS ----
#pragma unroll
        for (int j = 0; j < 2; ++j) {
            const float mv = mrow[key0 + j * 16 + lm];
#pragma unroll
            for (int r = 0; r < 8; ++r)
                P[(r + 8 * hi) * 33 + j * 16 + lm] = sc[j][r] * 0.125f + mv;
        }

        // ---- online softmax: 2 lanes per row (halved exp chain) ----
        {
            const int row = lm;
            float cmp = -1e30f;
#pragma unroll 4
            for (int c = 0; c < 16; ++c)
                cmp = fmaxf(cmp, P[row * 33 + hi * 16 + c]);
            pmax[hi * 16 + row] = cmp;
            const float om = rmax[row];
            const float cm = fmaxf(om, fmaxf(pmax[row], pmax[16 + row]));
            float sum = 0.0f;
#pragma unroll 4
            for (int c = 0; c < 16; ++c) {
                const float e = __expf(P[row * 33 + hi * 16 + c] - cm);
                sum += e;
                P[row * 33 + hi * 16 + c] = e;
            }
            psum[hi * 16 + row] = sum;
            if (lane < 16) {
                const float scale = __expf(om - cm);
                rsum[row]   = rsum[row] * scale + psum[row] + psum[16 + row];
                rmax[row]   = cm;
                rscale[row] = scale;
            }
        }

        // ---- rescale running context accumulators ----
#pragma unroll
        for (int r = 0; r < 8; ++r) {
            const float f = rscale[r + 8 * hi];
            o0[r] *= f; o1[r] *= f; o2[r] *= f; o3[r] *= f;
        }

        // ---- rebuild exp'd P tile as bf16 A-fragment (k = 32 keys) ----
        V16BF pa;
#pragma unroll
        for (int e = 0; e < 16; ++e) {
            const int kl = 16 * hi + e;
            pa.s[e] = f32_to_bf16(P[lm * 33 + kl]);
        }

        // ---- ctx += P * V  (4 dh tiles of 16) from LDS Vt chunk ----
        {
            V16BF vb;
            const uint4* vp;
            vp = (const uint4*)&Vs[(0 * 16 + lm) * 32 + 16 * hi];
            vb.q[0] = vp[0]; vb.q[1] = vp[1];
            o0 = __builtin_amdgcn_wmma_f32_16x16x32_bf16(
                     false, pa.v, false, vb.v, (short)0, o0, false, false);
            vp = (const uint4*)&Vs[(1 * 16 + lm) * 32 + 16 * hi];
            vb.q[0] = vp[0]; vb.q[1] = vp[1];
            o1 = __builtin_amdgcn_wmma_f32_16x16x32_bf16(
                     false, pa.v, false, vb.v, (short)0, o1, false, false);
            vp = (const uint4*)&Vs[(2 * 16 + lm) * 32 + 16 * hi];
            vb.q[0] = vp[0]; vb.q[1] = vp[1];
            o2 = __builtin_amdgcn_wmma_f32_16x16x32_bf16(
                     false, pa.v, false, vb.v, (short)0, o2, false, false);
            vp = (const uint4*)&Vs[(3 * 16 + lm) * 32 + 16 * hi];
            vb.q[0] = vp[0]; vb.q[1] = vp[1];
            o3 = __builtin_amdgcn_wmma_f32_16x16x32_bf16(
                     false, pa.v, false, vb.v, (short)0, o3, false, false);
        }

        __syncthreads();   // chunk fully consumed before next overwrite
    }

    // ---- normalize and write out [B, S, H*64] fp32 ----
#pragma unroll
    for (int r = 0; r < 8; ++r) {
        const int row = r + 8 * hi;
        const float inv = 1.0f / rsum[row];
        float* orow = out + ((size_t)(bb * Ss + q0 + row)) * Dd + h * DH;
        orow[0 * 16 + lm] = o0[r] * inv;
        orow[1 * 16 + lm] = o1[r] * inv;
        orow[2 * 16 + lm] = o2[r] * inv;
        orow[3 * 16 + lm] = o3[r] * inv;
    }
}

// ---------------------------------------------------------------------------
extern "C" void kernel_launch(void* const* d_in, const int* in_sizes, int n_in,
                              void* d_out, int out_size, void* d_ws, size_t ws_size,
                              hipStream_t stream) {
    const float* hidden = (const float*)d_in[0];   // [B,S,D] fp32
    const float* amask  = (const float*)d_in[1];   // [B,1,1,S] fp32
    const float* wqkv   = (const float*)d_in[2];   // [D, 3D] fp32
    float* out = (float*)d_out;

    char* ws = (char*)d_ws;
    unsigned short* hbf = (unsigned short*)(ws);                         // 16 MB
    unsigned short* wtb = (unsigned short*)(ws + (size_t)16777216);      //  6 MB
    unsigned short* qb  = (unsigned short*)(ws + (size_t)23068672);      // 16 MB
    unsigned short* kb  = (unsigned short*)(ws + (size_t)39845888);      // 16 MB
    unsigned short* vb  = (unsigned short*)(ws + (size_t)56623104);      // 16 MB

    k_cvt_hidden<<<8192, 256, 0, stream>>>(hidden, hbf);
    k_cvt_weight<<<dim3(Dd / 32, N3D / 32), 256, 0, stream>>>(wqkv, wtb);
    // 256 x 48 tiles of 32x64, 8 waves/block
    k_qkv_gemm<<<(256 * 48) / 8, 256, 0, stream>>>(hbf, wtb, qb, kb, vb);
    // 1024 blocks: 8 q-tile waves sharing one head's K/V chunks via LDS
    k_attention<<<1024, 256, 0, stream>>>(qb, kb, vb, amask, out);

    (void)in_sizes; (void)n_in; (void)out_size; (void)ws_size;
}
